// NeuralTensorDiagLayer_52321291600495
// MI455X (gfx1250) — matrix operand order repaired and checked by reading, not compiled, not run on hardware
//
#include <hip/hip_runtime.h>
#include <hip/hip_bf16.h>
#include <math.h>

typedef __bf16 bf16_t;
typedef __attribute__((ext_vector_type(16))) __bf16 v16bf;
typedef __attribute__((ext_vector_type(8)))  __bf16 v8bf;
typedef __attribute__((ext_vector_type(8)))  float  v8f;

#define BATCH 32768
#define DDIM  1024      // d
#define KTOT  2048      // 2d
#define NOUT  128       // k
#define BM    128       // M tile per block
#define KCH   32        // K chunk (per half) per iteration == WMMA K
#define LDA   40        // LDS row stride in bf16 (80B: 16B-aligned, bank-conflict-free)

// ---------------------------------------------------------------------------
// Kernel 0: zero the global column-product accumulator
// ---------------------------------------------------------------------------
__global__ void ntd_zero(float* __restrict__ p, int n) {
    int i = blockIdx.x * blockDim.x + threadIdx.x;
    if (i < n) p[i] = 0.0f;
}

// ---------------------------------------------------------------------------
// Kernel 1: transpose V [2048,128] fp32 -> Vt [128,2048] bf16 (WMMA B layout
// becomes 16 contiguous bf16 per lane). 512 KB, L2-resident, reused by all
// GEMM blocks.
// ---------------------------------------------------------------------------
__global__ void ntd_transpose_v(const float* __restrict__ V, bf16_t* __restrict__ Vt) {
    int idx = blockIdx.x * blockDim.x + threadIdx.x;   // 0 .. 2048*128-1
    int kk  = idx >> 7;          // 0..2047 (row of V)  -> coalesced read
    int n   = idx & 127;         // 0..127  (col of V)
    Vt[(size_t)n * KTOT + kk] = (bf16_t)V[(size_t)kk * NOUT + n];
}

// ---------------------------------------------------------------------------
// Kernel 2: fused GEMM (bf16 WMMA, f32 accum) + e1*e2 column-sum reduction.
// Block: 256 threads = 8 waves (4 M x 2 N), tile 128x128, K-loop over paired
// (e1,e2) chunks of 32 columns so both operands of the diag product are live.
// ---------------------------------------------------------------------------
__global__ __launch_bounds__(256)
void ntd_gemm(const float* __restrict__ e1, const float* __restrict__ e2,
              const bf16_t* __restrict__ Vt, float* __restrict__ out,
              float* __restrict__ colsum)
{
    __shared__ bf16_t sA1[BM * LDA];
    __shared__ bf16_t sA2[BM * LDA];
    __shared__ float  sCol[DDIM];

    const int tid   = threadIdx.x;
    const int wid   = tid >> 5;
    const int lane  = tid & 31;
    const int m0    = blockIdx.x * BM;
    const int waveM = wid & 3;       // 0..3 -> 32-row slab
    const int waveN = wid >> 2;      // 0..1 -> 64-col slab

    for (int j = tid; j < DDIM; j += 256) sCol[j] = 0.0f;

    v8f acc[2][4];
    #pragma unroll
    for (int f = 0; f < 2; ++f)
        #pragma unroll
        for (int c = 0; c < 4; ++c) acc[f][c] = (v8f){};

    const int lrow  = tid >> 3;        // row within a 32-row sweep group
    const int lcol4 = (tid & 7) * 4;   // 4-col group within the 32-col chunk

    union Frag { v16bf v; struct { v8bf lo; v8bf hi; } h; };

    for (int kc = 0; kc < DDIM / KCH; ++kc) {
        __syncthreads();   // previous iteration's LDS reads done

        // ---- load phase: e1/e2 fp32 -> products + bf16 tiles in LDS ----
        float pacc[4] = {0.f, 0.f, 0.f, 0.f};
        const int colg = kc * KCH + lcol4;
        #pragma unroll
        for (int s = 0; s < 4; ++s) {
            const int row = s * 32 + lrow;
            const size_t off = (size_t)(m0 + row) * DDIM + colg;
            const float4 a = *(const float4*)(e1 + off);
            const float4 c = *(const float4*)(e2 + off);
            pacc[0] += a.x * c.x;  pacc[1] += a.y * c.y;
            pacc[2] += a.z * c.z;  pacc[3] += a.w * c.w;
            bf16_t* p1 = &sA1[row * LDA + lcol4];
            p1[0] = (bf16_t)a.x; p1[1] = (bf16_t)a.y;
            p1[2] = (bf16_t)a.z; p1[3] = (bf16_t)a.w;
            bf16_t* p2 = &sA2[row * LDA + lcol4];
            p2[0] = (bf16_t)c.x; p2[1] = (bf16_t)c.y;
            p2[2] = (bf16_t)c.z; p2[3] = (bf16_t)c.w;
        }
        // wave-level row reduction (lanes l, l^8, l^16 share columns)
        #pragma unroll
        for (int c = 0; c < 4; ++c) {
            pacc[c] += __shfl_xor(pacc[c], 8, 32);
            pacc[c] += __shfl_xor(pacc[c], 16, 32);
        }
        if (lane < 8) {
            #pragma unroll
            for (int c = 0; c < 4; ++c)
                atomicAdd(&sCol[kc * KCH + lane * 4 + c], pacc[c]);
        }
        __syncthreads();   // tiles visible to all waves

        // ---- fragment loads ----
        v16bf a1f[2], a2f[2];
        const int akh = (lane >> 4) * 16;   // K sub-offset: contiguous k = 16*(lane>>4) + s
        #pragma unroll
        for (int f = 0; f < 2; ++f) {
            const int r = waveM * 32 + f * 16 + (lane & 15);
            Frag u;
            u.h.lo = *(const v8bf*)&sA1[r * LDA + akh];
            u.h.hi = *(const v8bf*)&sA1[r * LDA + 8 + akh];
            a1f[f] = u.v;
            u.h.lo = *(const v8bf*)&sA2[r * LDA + akh];
            u.h.hi = *(const v8bf*)&sA2[r * LDA + 8 + akh];
            a2f[f] = u.v;
        }
        v16bf b1f[4], b2f[4];
        const int bkh = (lane >> 4) * 16;   // K sub-offset: same contiguous k order as A
        #pragma unroll
        for (int c = 0; c < 4; ++c) {
            const int n = waveN * 64 + c * 16 + (lane & 15);
            b1f[c] = *(const v16bf*)(Vt + (size_t)n * KTOT + kc * KCH + bkh);
            b2f[c] = *(const v16bf*)(Vt + (size_t)n * KTOT + DDIM + kc * KCH + bkh);
        }

        // ---- 16 WMMAs per iteration per wave ----
        #pragma unroll
        for (int f = 0; f < 2; ++f)
            #pragma unroll
            for (int c = 0; c < 4; ++c) {
                acc[f][c] = __builtin_amdgcn_wmma_f32_16x16x32_bf16(
                    false, a1f[f], false, b1f[c], (short)0, acc[f][c], false, false);
                acc[f][c] = __builtin_amdgcn_wmma_f32_16x16x32_bf16(
                    false, a2f[f], false, b2f[c], (short)0, acc[f][c], false, false);
            }
    }

    // ---- store pre-activation (C layout: n = lane&15, m = (lane>=16)*8 + v) ----
    #pragma unroll
    for (int f = 0; f < 2; ++f)
        #pragma unroll
        for (int c = 0; c < 4; ++c) {
            const int n = waveN * 64 + c * 16 + (lane & 15);
            const int mbase = m0 + waveM * 32 + f * 16 + (lane >> 4) * 8;
            #pragma unroll
            for (int v = 0; v < 8; ++v)
                out[(size_t)(mbase + v) * NOUT + n] = acc[f][c][v];
        }

    // ---- flush block's column-product sums ----
    __syncthreads();
    for (int j = tid; j < DDIM; j += 256) atomicAdd(colsum + j, sCol[j]);
}

// ---------------------------------------------------------------------------
// Kernel 3: diag[i] + b[i]; one block per output i.
// ---------------------------------------------------------------------------
__global__ __launch_bounds__(256)
void ntd_diag(const float* __restrict__ W, const float* __restrict__ b,
              const float* __restrict__ colsum, float* __restrict__ dpb)
{
    __shared__ float red[256];
    const int i = blockIdx.x;
    float s = 0.0f;
    for (int j = threadIdx.x; j < DDIM; j += 256)
        s += W[(size_t)i * DDIM + j] * colsum[j];
    red[threadIdx.x] = s;
    __syncthreads();
    for (int o = 128; o > 0; o >>= 1) {
        if (threadIdx.x < o) red[threadIdx.x] += red[threadIdx.x + o];
        __syncthreads();
    }
    if (threadIdx.x == 0)
        dpb[i] = b[i] + red[0] * (1.0f / ((float)DDIM * (float)BATCH));
}

// ---------------------------------------------------------------------------
// Kernel 4: out = tanh(out + dpb[n]) — vectorized epilogue.
// ---------------------------------------------------------------------------
__global__ __launch_bounds__(256)
void ntd_tanh(float* __restrict__ out, const float* __restrict__ dpb)
{
    __shared__ float sd[NOUT];
    if (threadIdx.x < NOUT) sd[threadIdx.x] = dpb[threadIdx.x];
    __syncthreads();
    const size_t i4 = ((size_t)blockIdx.x * 256 + threadIdx.x) * 4;
    float4 v = *(float4*)(out + i4);
    const int n = (int)(i4 & (NOUT - 1));   // 4 | 128 -> never crosses a row
    v.x = tanhf(v.x + sd[n + 0]);
    v.y = tanhf(v.y + sd[n + 1]);
    v.z = tanhf(v.z + sd[n + 2]);
    v.w = tanhf(v.w + sd[n + 3]);
    *(float4*)(out + i4) = v;
}

// ---------------------------------------------------------------------------
extern "C" void kernel_launch(void* const* d_in, const int* in_sizes, int n_in,
                              void* d_out, int out_size, void* d_ws, size_t ws_size,
                              hipStream_t stream)
{
    const float* e1 = (const float*)d_in[0];
    const float* e2 = (const float*)d_in[1];
    const float* W  = (const float*)d_in[2];
    const float* V  = (const float*)d_in[3];
    const float* b  = (const float*)d_in[4];
    float* out = (float*)d_out;

    char* ws = (char*)d_ws;
    bf16_t* Vt     = (bf16_t*)ws;                       // 128*2048*2 = 524288 B
    float*  colsum = (float*)(ws + 524288);             // 1024*4    =   4096 B
    float*  dpb    = (float*)(ws + 524288 + 4096);      // 128*4     =    512 B

    ntd_zero<<<(DDIM + 255) / 256, 256, 0, stream>>>(colsum, DDIM);
    ntd_transpose_v<<<(KTOT * NOUT) / 256, 256, 0, stream>>>(V, Vt);
    ntd_gemm<<<BATCH / BM, 256, 0, stream>>>(e1, e2, Vt, out, colsum);
    ntd_diag<<<NOUT, 256, 0, stream>>>(W, b, colsum, dpb);
    ntd_tanh<<<(BATCH * NOUT) / 1024, 256, 0, stream>>>(out, dpb);
}
